// SwinGraphTransformer_28097676050906
// MI455X (gfx1250) — matrix operand order repaired and checked by reading, not compiled, not run on hardware
//
#include <hip/hip_runtime.h>
#include <hip/hip_bf16.h>

typedef __attribute__((ext_vector_type(16))) _Float16 v16h;
typedef __attribute__((ext_vector_type(8)))  _Float16 v8h;
typedef __attribute__((ext_vector_type(8)))  float    v8f;

#define DH 64

// ---------------- block reduction helpers (blockDim must be pow2) ----------------
__device__ __forceinline__ float blkSum(float v, float* sh) {
    int t = threadIdx.x;
    sh[t] = v; __syncthreads();
    for (int o = blockDim.x >> 1; o > 0; o >>= 1) {
        if (t < o) sh[t] += sh[t + o];
        __syncthreads();
    }
    float r = sh[0]; __syncthreads();
    return r;
}
__device__ __forceinline__ float blkMax(float v, float* sh) {
    int t = threadIdx.x;
    sh[t] = v; __syncthreads();
    for (int o = blockDim.x >> 1; o > 0; o >>= 1) {
        if (t < o) sh[t] = fmaxf(sh[t], sh[t + o]);
        __syncthreads();
    }
    float r = sh[0]; __syncthreads();
    return r;
}

// A fragment: 16x32 f16: lanes<16 elems0..15=K0..15;
// lanes>=16 elems0..15=K16..31. A is K-contiguous -> float4 loads.
__device__ __forceinline__ v16h loadAfrag(const float* __restrict__ Ab, int k0, int hi) {
    v16h af;
    #pragma unroll
    for (int c = 0; c < 2; ++c) {
        const float* ap = Ab + k0 + hi * 16 + c * 8;
        float4 v0 = *(const float4*)(ap);
        float4 v1 = *(const float4*)(ap + 4);
        af[c * 8 + 0] = (_Float16)v0.x; af[c * 8 + 1] = (_Float16)v0.y;
        af[c * 8 + 2] = (_Float16)v0.z; af[c * 8 + 3] = (_Float16)v0.w;
        af[c * 8 + 4] = (_Float16)v1.x; af[c * 8 + 5] = (_Float16)v1.y;
        af[c * 8 + 6] = (_Float16)v1.z; af[c * 8 + 7] = (_Float16)v1.w;
    }
    return af;
}

// ---------------- generic strided batched WMMA GEMM (aligned shapes) ----------------
// REQUIRES: M%16==0, Kd%32==0, N%16==0. Either sBk==1 (K-contiguous B) or sBn==1.
// C[m,n] = act( alpha * sum_k A[m,k]*B[k,n] + bias[n] )
// A elem at A[offA + m*sAm + k]; B elem at B[offB + k*sBk + n*sBn]; C at C[offC + m*sCm + n].
// batch b -> (b/hCount, b%hCount) with strides (aKs,aHs)/(bKs,bHs)/(cKs,cHs).
// 4 waves/block, block tile M16 x N64; each wave32 -> one 16x16 tile via v_wmma_f32_16x16x32_f16.
__global__ void k_gemm(const float* __restrict__ A, const float* __restrict__ B,
                       float* __restrict__ C, const float* __restrict__ bias,
                       int M, int N, int Kd,
                       long sAm, long sBk, long sBn, long sCm,
                       int hCount,
                       long aKs, long aHs, long bKs, long bHs, long cKs, long cHs,
                       int act, float alpha) {
    __shared__ _Float16 Bs[64 * 40];     // B slab transposed: Bs[colLocal*40 + kLocal]
    int lane = threadIdx.x & 31;
    int wave = threadIdx.x >> 5;
    int b    = blockIdx.z;
    int bk = b / hCount, bh = b % hCount;
    long offA = (long)bk * aKs + (long)bh * aHs;
    long offB = (long)bk * bKs + (long)bh * bHs;
    long offC = (long)bk * cKs + (long)bh * cHs;
    int m0   = blockIdx.y * 16;
    int nblk = blockIdx.x * 64;
    int n0   = nblk + wave * 16;
    int row  = lane & 15;
    int hi   = lane >> 4;
    const float* Ab = A + offA + (long)(m0 + row) * sAm;   // K-contiguous
    v8f acc = {};

    if (sBk == 1) {
        // ---- direct path (q.k^T): B K-contiguous, float4 loads, no LDS ----
        if (n0 >= N) return;             // wave-uniform, EXEC stays all-ones for WMMA
        const float* Bb = B + offB + (long)(n0 + row) * sBn;
        for (int k0 = 0; k0 < Kd; k0 += 32) {
            if (k0 + 32 < Kd) {
                __builtin_prefetch(&Ab[k0 + 32], 0, 1);
                __builtin_prefetch(&Bb[k0 + 32], 0, 1);
            }
            v16h af = loadAfrag(Ab, k0, hi);
            v16h bf;
            const float* bp = Bb + k0 + hi * 16;
            #pragma unroll
            for (int c = 0; c < 4; ++c) {
                float4 v0 = *(const float4*)(bp + c * 4);
                bf[c * 4 + 0] = (_Float16)v0.x; bf[c * 4 + 1] = (_Float16)v0.y;
                bf[c * 4 + 2] = (_Float16)v0.z; bf[c * 4 + 3] = (_Float16)v0.w;
            }
            acc = __builtin_amdgcn_wmma_f32_16x16x32_f16(false, af, false, bf,
                                                         (short)0, acc, false, false);
        }
    } else {
        // ---- staged path (weights, a.v): sBn==1, rows N-contiguous.
        // Block cooperatively loads 32x64 slab with coalesced float4s, stores f16
        // transposed into LDS; lanes then read 16-element K-runs as 16B ds loads.
        for (int k0 = 0; k0 < Kd; k0 += 32) {
            __syncthreads();
            #pragma unroll
            for (int i = 0; i < 4; ++i) {
                int idx  = threadIdx.x + 128 * i;   // 512 float4s = 32 rows x 16
                int krow = idx >> 4;
                int c4   = (idx & 15) << 2;
                int col  = nblk + c4;
                float4 v = make_float4(0.f, 0.f, 0.f, 0.f);
                if (col < N)
                    v = *(const float4*)&B[offB + (long)(k0 + krow) * sBk + col];
                Bs[(c4 + 0) * 40 + krow] = (_Float16)v.x;
                Bs[(c4 + 1) * 40 + krow] = (_Float16)v.y;
                Bs[(c4 + 2) * 40 + krow] = (_Float16)v.z;
                Bs[(c4 + 3) * 40 + krow] = (_Float16)v.w;
            }
            __syncthreads();
            if (k0 + 32 < Kd) __builtin_prefetch(&Ab[k0 + 32], 0, 1);
            v16h af = loadAfrag(Ab, k0, hi);
            v16h bf;
            const _Float16* bp = &Bs[(wave * 16 + row) * 40 + hi * 16];
            v8h b0 = *(const v8h*)(bp);
            v8h b1 = *(const v8h*)(bp + 8);
            #pragma unroll
            for (int t = 0; t < 8; ++t) { bf[t] = b0[t]; bf[8 + t] = b1[t]; }
            acc = __builtin_amdgcn_wmma_f32_16x16x32_f16(false, af, false, bf,
                                                         (short)0, acc, false, false);
        }
        if (n0 >= N) return;             // safe: no more barriers below
    }
    // C/D layout: VGPR r -> (lanes<16: M=r, lanes>=16: M=8+r), col = lane&15.
    int col = n0 + row;
    float bv = bias ? bias[col] : 0.0f;
    #pragma unroll
    for (int r = 0; r < 8; ++r) {
        int mr = m0 + (hi ? 8 + r : r);
        float v = acc[r] * alpha + bv;
        if (act == 1) v = 0.5f * v * (1.0f + erff(v * 0.70710678118f)); // exact GELU
        C[offC + (long)mr * sCm + col] = v;
    }
}

// ---------------- elementwise / row kernels ----------------
__global__ void k_layernorm(const float* __restrict__ x, float* __restrict__ y,
                            const float* __restrict__ g, const float* __restrict__ b, int dim) {
    __shared__ float sh[128];
    long base = (long)blockIdx.x * dim;
    float s = 0.0f;
    for (int i = threadIdx.x; i < dim; i += blockDim.x) s += x[base + i];
    float mean = blkSum(s, sh) / (float)dim;
    float v = 0.0f;
    for (int i = threadIdx.x; i < dim; i += blockDim.x) { float d = x[base + i] - mean; v += d * d; }
    float var = blkSum(v, sh) / (float)dim;
    float inv = rsqrtf(var + 1e-5f);
    for (int i = threadIdx.x; i < dim; i += blockDim.x)
        y[base + i] = (x[base + i] - mean) * inv * g[i] + b[i];
}

// RoPE in place on [K*n, rowStride] where row layout is [h*DH + d]; pos = node % n.
__global__ void k_rope(float* __restrict__ t, int n, int heads, int rowStride, int totalPairs) {
    int p = blockIdx.x * blockDim.x + threadIdx.x;
    if (p >= totalPairs) return;
    int d2   = p & 31;                 // DH/2 = 32 pairs
    int h    = (p >> 5) % heads;
    int node = p / (32 * heads);
    int pos  = node % n;
    float inv = expf(-9.2103403720f * (float)(2 * d2) * (1.0f / 64.0f)); // 10000^{-2i/64}
    float f = (float)pos * inv;
    float c = cosf(f), s = sinf(f);
    long base = (long)node * rowStride + h * DH + 2 * d2;
    float t0 = t[base], t1 = t[base + 1];
    t[base]     = t0 * c - t1 * s;
    t[base + 1] = t1 * c + t0 * s;
}

// softmax over j with edge bias: logit = (sim + (adj? q.r1 : q.r0)) * scale
// also emits s = sum_j a*adj. One block of 128 threads per (kk,h,i) row; n == 128.
__global__ void k_softmax_adj(float* __restrict__ sim, const float* __restrict__ q,
                              const float* __restrict__ r, const int* __restrict__ adj,
                              float* __restrict__ sgl, int n, int heads, int inner, float scale) {
    __shared__ float sh[128];
    int b  = blockIdx.x;                 // (kk*heads + h)*n + i
    int i  = b % n;
    int h  = (b / n) % heads;
    int kk = b / (n * heads);
    int t  = threadIdx.x;

    const float* qrow = q + ((long)(kk * n + i) * inner + h * DH);
    float p0 = 0.0f, p1 = 0.0f;
    if (t < DH) {
        float qv = qrow[t];
        p0 = qv * r[h * DH + t];
        p1 = qv * r[inner + h * DH + t];
    }
    float qr0 = blkSum(p0, sh);
    float qr1 = blkSum(p1, sh);

    float* rowp = sim + (long)b * n;
    const int* arow = adj + (long)(kk * n + i) * n;
    int aj = 0;
    float lg = -1e30f;
    if (t < n) {
        aj = arow[t];
        lg = (rowp[t] + (aj ? qr1 : qr0)) * scale;
    }
    float m  = blkMax(lg, sh);
    float ex = (t < n) ? expf(lg - m) : 0.0f;
    float ss = blkSum(ex, sh);
    float a  = ex / ss;
    if (t < n) rowp[t] = a;
    float sA = blkSum((t < n && aj) ? a : 0.0f, sh);
    if (t == 0) sgl[b] = sA;
}

// av[node, h*DH+d] += (1-s)*r0[h,d] + s*r1[h,d]
__global__ void k_edge_out_add(float* __restrict__ av, const float* __restrict__ sgl,
                               const float* __restrict__ r, int n, int heads, int inner, int total) {
    int t = blockIdx.x * blockDim.x + threadIdx.x;
    if (t >= total) return;
    int d    = t & (DH - 1);
    int h    = (t / DH) % heads;
    int node = t / (DH * heads);
    int kk = node / n, i = node % n;
    float s = sgl[((long)kk * heads + h) * n + i];
    av[t] += (1.0f - s) * r[h * DH + d] + s * r[inner + h * DH + d];
}

// out = h*g + res*(1-g); g = sigmoid([h,res,h-res] . w). out may alias res.
__global__ void k_gated_res(const float* __restrict__ h, const float* __restrict__ res,
                            const float* __restrict__ w, float* __restrict__ out, int dim) {
    __shared__ float sh[128];
    long base = (long)blockIdx.x * dim;
    float acc = 0.0f;
    for (int i = threadIdx.x; i < dim; i += blockDim.x) {
        float hv = h[base + i], rv = res[base + i];
        acc += hv * w[i] + rv * w[dim + i] + (hv - rv) * w[2 * dim + i];
    }
    float z = blkSum(acc, sh);
    float g = 1.0f / (1.0f + expf(-z));
    for (int i = threadIdx.x; i < dim; i += blockDim.x)
        out[base + i] = h[base + i] * g + res[base + i] * (1.0f - g);
}

__global__ void k_add(float* __restrict__ y, const float* __restrict__ x, int total) {
    int t = blockIdx.x * blockDim.x + threadIdx.x;
    if (t < total) y[t] += x[t];
}
__global__ void k_roll_gather(const float* __restrict__ x, float* __restrict__ out,
                              int N, int dim, int shift) {
    int t = blockIdx.x * blockDim.x + threadIdx.x;
    if (t >= N * dim) return;
    int i = t / dim, c = t % dim;
    out[t] = x[(long)((i + shift) % N) * dim + c];
}
__global__ void k_roll_scatter_add(float* __restrict__ x, const float* __restrict__ out,
                                   int N, int dim, int shift) {
    int t = blockIdx.x * blockDim.x + threadIdx.x;
    if (t >= N * dim) return;
    int i = t / dim, c = t % dim;
    x[(long)((i + shift) % N) * dim + c] += out[t];
}

__global__ void k_build_adj(const int* __restrict__ src, const int* __restrict__ dst,
                            int* __restrict__ adj, int E, int n, int shift, int N) {
    int e = blockIdx.x * blockDim.x + threadIdx.x;
    if (e >= E) return;
    int s = src[e], d = dst[e];
    if (s < 0 || d < 0) return;
    int ss = (s - shift) % N; if (ss < 0) ss += N;
    int dd = (d - shift) % N; if (dd < 0) dd += N;
    int cs = ss / n, cd = dd / n;
    if (cs != cd) return;
    adj[(long)cs * n * n + (ss % n) * n + (dd % n)] = 1;
}

__global__ void k_score(const float* __restrict__ x, const float* __restrict__ w,
                        float* __restrict__ score, int dim) {
    __shared__ float sh[128];
    long base = (long)blockIdx.x * dim;
    float d = 0.0f, ws = 0.0f;
    for (int i = threadIdx.x; i < dim; i += blockDim.x) {
        float wv = w[i];
        d  += x[base + i] * wv;
        ws += wv * wv;
    }
    float dt = blkSum(d, sh);
    float wt = blkSum(ws, sh);
    if (threadIdx.x == 0) score[blockIdx.x] = tanhf(dt * rsqrtf(wt));
}

// single-block bitonic sort, descending by value (ties: lower index first). N pow2 <= 1024.
__global__ void k_topk_sort(const float* __restrict__ score, int* __restrict__ perm,
                            float* __restrict__ vals, int N) {
    __shared__ float sv[1024];
    __shared__ int   si[1024];
    for (int i = threadIdx.x; i < N; i += blockDim.x) { sv[i] = score[i]; si[i] = i; }
    __syncthreads();
    for (int ksz = 2; ksz <= N; ksz <<= 1) {
        for (int j = ksz >> 1; j > 0; j >>= 1) {
            for (int i = threadIdx.x; i < N; i += blockDim.x) {
                int ixj = i ^ j;
                if (ixj > i) {
                    float a = sv[i], bb = sv[ixj];
                    int ia = si[i], ib = si[ixj];
                    bool aFirst = (a > bb) || (a == bb && ia < ib);
                    bool descSeg = ((i & ksz) == 0);
                    if (descSeg ? !aFirst : aFirst) {
                        sv[i] = bb; si[i] = ib; sv[ixj] = a; si[ixj] = ia;
                    }
                }
            }
            __syncthreads();
        }
    }
    for (int i = threadIdx.x; i < N; i += blockDim.x) { perm[i] = si[i]; vals[i] = sv[i]; }
}

__global__ void k_pool_apply(const float* __restrict__ xin, const int* __restrict__ perm,
                             const float* __restrict__ vals, float* __restrict__ xout,
                             int k, int dim) {
    int t = blockIdx.x * blockDim.x + threadIdx.x;
    if (t >= k * dim) return;
    int i = t / dim, c = t % dim;
    xout[t] = xin[(long)perm[i] * dim + c] * vals[i];
}
__global__ void k_node_map(int* __restrict__ nmap, const int* __restrict__ perm, int k) {
    int i = blockIdx.x * blockDim.x + threadIdx.x;
    if (i < k) nmap[perm[i]] = i;
}
__global__ void k_remap(int* __restrict__ e, const int* __restrict__ nmap, int E, int N) {
    int i = blockIdx.x * blockDim.x + threadIdx.x;
    if (i >= E) return;
    int v = e[i];
    int u = v < 0 ? 0 : (v > N - 1 ? N - 1 : v);
    e[i] = (v >= 0) ? nmap[u] : -1;
}
__global__ void k_perm_gather(int* __restrict__ pg2, const int* __restrict__ pg,
                              const int* __restrict__ perm, int k) {
    int i = blockIdx.x * blockDim.x + threadIdx.x;
    if (i < k) pg2[i] = pg[perm[i]];
}
__global__ void k_iota(int* __restrict__ p, int n) {
    int i = blockIdx.x * blockDim.x + threadIdx.x;
    if (i < n) p[i] = i;
}
__global__ void k_write_perm(int* __restrict__ outp, const int* __restrict__ permg, int n) {
    int i = blockIdx.x * blockDim.x + threadIdx.x;
    if (i < n) outp[i] = permg[i];
}

// ---------------- host-side param structs ----------------
struct AttnP { const float *be,*bkv,*bo,*bq,*gate,*ln_b,*ln_g,*we,*wkv,*wo,*wq; };
struct FfP   { const float *b1,*b2,*gate,*ln_b,*ln_g,*w1,*w2; };
struct GtP   { const float* adj_emb; AttnP attn[2]; FfP ff[2]; };
struct StageP{ const float *exp_b,*exp_w,*pool_w; GtP shiftGt, winGt; };

extern "C" void kernel_launch(void* const* d_in, const int* in_sizes, int n_in,
                              void* d_out, int out_size, void* d_ws, size_t ws_size,
                              hipStream_t stream) {
    (void)n_in; (void)out_size; (void)ws_size;
    const float* x_in = (const float*)d_in[0];
    const int*   ei   = (const int*)d_in[1];
    const int E = in_sizes[1] / 2;

    // ---- params: jax tree_leaves order (dict keys sorted, lists in order) ----
    int idx = 2;
    auto nx = [&]() { return (const float*)d_in[idx++]; };
    const float* proj_b  = nx();
    const float* proj_w  = nx();
    const float* ro_b    = nx();
    const float* ro_ln_b = nx();
    const float* ro_ln_g = nx();
    const float* ro_w    = nx();
    StageP st[2];
    for (int s = 0; s < 2; ++s) {
        st[s].exp_b = nx(); st[s].exp_w = nx(); st[s].pool_w = nx();
        auto readGt = [&](GtP& g) {
            g.adj_emb = nx();
            for (int L = 0; L < 2; ++L) {
                AttnP& a = g.attn[L];
                a.be = nx(); a.bkv = nx(); a.bo = nx(); a.bq = nx(); a.gate = nx();
                a.ln_b = nx(); a.ln_g = nx(); a.we = nx(); a.wkv = nx(); a.wo = nx(); a.wq = nx();
                FfP& f = g.ff[L];
                f.b1 = nx(); f.b2 = nx(); f.gate = nx(); f.ln_b = nx(); f.ln_g = nx();
                f.w1 = nx(); f.w2 = nx();
            }
        };
        readGt(st[s].shiftGt);
        readGt(st[s].winGt);
    }

    // ---- workspace bump allocator ----
    size_t off = 0;
    auto allocf = [&](size_t n) -> float* {
        size_t a = (off + 255) & ~(size_t)255; off = a + n * sizeof(float);
        return (float*)((char*)d_ws + a);
    };
    auto alloci = [&](size_t n) -> int* {
        size_t a = (off + 255) & ~(size_t)255; off = a + n * sizeof(int);
        return (int*)((char*)d_ws + a);
    };
    float* X    = allocf(393216);
    float* Xw   = allocf(393216);
    float* X2   = allocf(393216);
    float* ln   = allocf(393216);
    float* h1   = allocf(131072);
    float* h2   = allocf(393216);
    float* qb   = allocf(262144);
    float* kvb  = allocf(524288);
    float* simb = allocf(524288);
    float* avb  = allocf(262144);
    float* rb   = allocf(8192);     // 16 x inner (only rows 0,1 meaningful)
    float* aePad= allocf(3072);     // 16 x dim zero-padded adj_emb
    float* sb   = allocf(4096);
    float* score= allocf(1024);
    float* valsb= allocf(1024);
    int* adj    = alloci(131072);
    int* srcb   = alloci(E);
    int* dstb   = alloci(E);
    int* permb  = alloci(1024);
    int* nmap   = alloci(1024);
    int* permg  = alloci(1024);
    int* permg2 = alloci(1024);

    // ---- GEMM launch helpers (M%16==0, K%32==0 guaranteed by caller) ----
    auto gemm2 = [&](const float* A, const float* Bm, float* Cm, const float* bias,
                     int M, int Nc, int Kd, int act) {
        dim3 g((Nc + 63) / 64, M / 16, 1);
        k_gemm<<<g, 128, 0, stream>>>(A, Bm, Cm, bias, M, Nc, Kd,
                                      (long)Kd, (long)Nc, 1L, (long)Nc,
                                      1, 0L, 0L, 0L, 0L, 0L, 0L, act, 1.0f);
    };
    auto gemmB = [&](const float* A, const float* Bm, float* Cm,
                     int M, int Nc, int Kd,
                     long sAm, long sBk, long sBn, long sCm,
                     int batches, int hCount,
                     long aKs, long aHs, long bKs, long bHs, long cKs, long cHs) {
        dim3 g((Nc + 63) / 64, M / 16, batches);
        k_gemm<<<g, 128, 0, stream>>>(A, Bm, Cm, nullptr, M, Nc, Kd,
                                      sAm, sBk, sBn, sCm,
                                      hCount, aKs, aHs, bKs, bHs, cKs, cHs, 0, 1.0f);
    };

    // ---- one GraphTransformer (depth=2) on clustered nodes Xn [Kc*n, dim] ----
    auto run_gt = [&](float* Xn, int Kc, int n, int dim, int heads, const GtP& gp) {
        int inner = heads * DH;
        int Nn = Kc * n;
        // zero-padded adj_emb rows (16 x dim) so the r-row GEMM is WMMA-aligned
        hipMemsetAsync(aePad, 0, (size_t)16 * dim * sizeof(float), stream);
        hipMemcpyAsync(aePad, gp.adj_emb, (size_t)2 * dim * sizeof(float),
                       hipMemcpyDeviceToDevice, stream);
        for (int L = 0; L < 2; ++L) {
            const AttnP& ap = gp.attn[L];
            const FfP&   fp = gp.ff[L];
            k_layernorm<<<Nn, 128, 0, stream>>>(Xn, ln, ap.ln_g, ap.ln_b, dim);
            gemm2(ln, ap.wq, qb, ap.bq, Nn, inner, dim, 0);
            gemm2(ln, ap.wkv, kvb, ap.bkv, Nn, 2 * inner, dim, 0);
            gemm2(aePad, ap.we, rb, ap.be, 16, inner, dim, 0);   // rows 0,1 = r0,r1
            int pairs = Nn * heads * (DH / 2);
            k_rope<<<(pairs + 255) / 256, 256, 0, stream>>>(qb, n, heads, inner, pairs);
            k_rope<<<(pairs + 255) / 256, 256, 0, stream>>>(kvb, n, heads, 2 * inner, pairs);
            // sim[k,h,i,j] = q_i . k_j   (batched over K*heads, B = k read transposed, sBk==1)
            gemmB(qb, kvb, simb, n, n, DH,
                  (long)inner, 1L, (long)(2 * inner), (long)n,
                  Kc * heads, heads,
                  (long)n * inner, (long)DH, (long)n * 2 * inner, (long)DH,
                  (long)heads * n * n, (long)n * n);
            k_softmax_adj<<<Kc * heads * n, 128, 0, stream>>>(simb, qb, rb, adj, sb,
                                                              n, heads, inner, 0.125f);
            // av[k,i,h,d] = sum_j a * v_j   (B strided rows -> LDS-staged path)
            gemmB(simb, kvb + inner, avb, n, DH, n,
                  (long)n, (long)(2 * inner), 1L, (long)inner,
                  Kc * heads, heads,
                  (long)heads * n * n, (long)n * n, (long)n * 2 * inner, (long)DH,
                  (long)n * inner, (long)DH);
            int tot = Nn * inner;
            k_edge_out_add<<<(tot + 255) / 256, 256, 0, stream>>>(avb, sb, rb, n, heads, inner, tot);
            gemm2(avb, ap.wo, h1, ap.bo, Nn, dim, inner, 0);
            k_gated_res<<<Nn, 128, 0, stream>>>(h1, Xn, ap.gate, Xn, dim);
            // FF
            k_layernorm<<<Nn, 128, 0, stream>>>(Xn, ln, fp.ln_g, fp.ln_b, dim);
            gemm2(ln, fp.w1, h2, fp.b1, Nn, 4 * dim, dim, 1);        // exact GELU fused
            gemm2(h2, fp.w2, h1, fp.b2, Nn, dim, 4 * dim, 0);
            k_gated_res<<<Nn, 128, 0, stream>>>(h1, Xn, fp.gate, Xn, dim);
        }
    };

    // ---- forward ----
    gemm2(x_in, proj_w, X, proj_b, 1024, 96, 128, 0);
    k_iota<<<4, 256, 0, stream>>>(permg, 1024);
    hipMemcpyAsync(srcb, ei,     (size_t)E * sizeof(int), hipMemcpyDeviceToDevice, stream);
    hipMemcpyAsync(dstb, ei + E, (size_t)E * sizeof(int), hipMemcpyDeviceToDevice, stream);

    int N = 1024;
    const int dims[3] = {96, 192, 384};
    const int KSs[2]  = {8, 4};
    const int HS[2]   = {4, 8};
    for (int s = 0; s < 2; ++s) {
        int Kc = KSs[s], n = N / Kc, shift = N / (2 * Kc), heads = HS[s];
        int din = dims[s], dout = dims[s + 1];
        // window block: x = x + GT(x)
        hipMemsetAsync(adj, 0, (size_t)Kc * n * n * sizeof(int), stream);
        k_build_adj<<<(E + 255) / 256, 256, 0, stream>>>(srcb, dstb, adj, E, n, 0, N);
        hipMemcpyAsync(Xw, X, (size_t)N * din * sizeof(float), hipMemcpyDeviceToDevice, stream);
        run_gt(Xw, Kc, n, din, heads, st[s].winGt);
        { int tot = N * din; k_add<<<(tot + 255) / 256, 256, 0, stream>>>(X, Xw, tot); }
        // shifted-window block: x[idx] += GT(x[idx])
        hipMemsetAsync(adj, 0, (size_t)Kc * n * n * sizeof(int), stream);
        k_build_adj<<<(E + 255) / 256, 256, 0, stream>>>(srcb, dstb, adj, E, n, shift, N);
        { int tot = N * din; k_roll_gather<<<(tot + 255) / 256, 256, 0, stream>>>(X, Xw, N, din, shift); }
        run_gt(Xw, Kc, n, din, heads, st[s].shiftGt);
        { int tot = N * din; k_roll_scatter_add<<<(tot + 255) / 256, 256, 0, stream>>>(X, Xw, N, din, shift); }
        // expand + TopK pooling (ratio 0.5)
        gemm2(X, st[s].exp_w, X2, st[s].exp_b, N, dout, din, 0);
        k_score<<<N, 128, 0, stream>>>(X2, st[s].pool_w, score, dout);
        int kk = N / 2;
        k_topk_sort<<<1, 512, 0, stream>>>(score, permb, valsb, N);
        { int tot = kk * dout;
          k_pool_apply<<<(tot + 255) / 256, 256, 0, stream>>>(X2, permb, valsb, X, kk, dout); }
        hipMemsetAsync(nmap, 0xFF, (size_t)N * sizeof(int), stream);
        k_node_map<<<(kk + 255) / 256, 256, 0, stream>>>(nmap, permb, kk);
        k_remap<<<(E + 255) / 256, 256, 0, stream>>>(srcb, nmap, E, N);
        k_remap<<<(E + 255) / 256, 256, 0, stream>>>(dstb, nmap, E, N);
        k_perm_gather<<<(kk + 255) / 256, 256, 0, stream>>>(permg2, permg, permb, kk);
        { int* tmp = permg; permg = permg2; permg2 = tmp; }
        N = kk;
    }
    // readout: y = LN(x) @ ro_w + ro_b   -> d_out[0 : 256*384]
    k_layernorm<<<N, 128, 0, stream>>>(X, ln, ro_ln_g, ro_ln_b, 384);
    gemm2(ln, ro_w, (float*)d_out, ro_b, N, 384, 384, 0);
    // perm_global ints appended after y
    k_write_perm<<<1, 256, 0, stream>>>((int*)d_out + 256 * 384, permg, 256);
}
